// StructureLayer_28381143892865
// MI455X (gfx1250) — compile-verified, matrix operand order checked by reading, not run on hardware
//
#include <hip/hip_runtime.h>
#include <math.h>

// ---------------------------------------------------------------------------
// StructureLayer (IPA + transition + frame update) for MI455X / gfx1250
// All heavy GEMMs use v_wmma_f32_16x16x32_bf16 (bf16 in, f32 accumulate).
// Register-blocked: one wave computes a (16*BM x 16*BN) output tile.
// ---------------------------------------------------------------------------

#define NRES 1024
#define CSD  384
#define CPD  128
#define CHD  16
#define NHD  12
#define PQD  4
#define PVD  8
#define NNL  (NRES * NRES)
#define PROJC 1152          // 192(q) + 384(kv) + 144(qp) + 432(kvp)
#define CATC  2112          // 192 + 96*4 + 1536

typedef __bf16 bf16_t;
typedef __attribute__((ext_vector_type(16))) __bf16 v16bf;
typedef __attribute__((ext_vector_type(8)))  __bf16 v8bf;
typedef __attribute__((ext_vector_type(8)))  float  v8f;

// ------------------------- WMMA fragment helpers ---------------------------
// 16-bit A/B 16x32 fragment layout (CDNA5 ISA 7.12.2):
//   lane<16 : K = {0..7, 16..23};  lane>=16 : K = {8..15, 24..31}
// loader takes kbase = k0 + (lane>>4)*8 and fetches [kbase..kbase+7] and
// [kbase+16..kbase+23] from a row-major row.
__device__ __forceinline__ v16bf frag_bf(const bf16_t* row, int kbase) {
  v8bf lo = *reinterpret_cast<const v8bf*>(row + kbase);
  v8bf hi = *reinterpret_cast<const v8bf*>(row + kbase + 16);
  v16bf f;
#pragma unroll
  for (int e = 0; e < 8; ++e) { f[e] = lo[e]; f[8 + e] = hi[e]; }
  return f;
}

__device__ __forceinline__ v16bf frag_f32(const float* row, int kbase) {
  v16bf f;
#pragma unroll
  for (int e = 0; e < 8; ++e) {
    f[e]     = (bf16_t)row[kbase + e];
    f[8 + e] = (bf16_t)row[kbase + 16 + e];
  }
  return f;
}

__device__ __forceinline__ v16bf frag_f32_strided(const float* base, int kbase, int stride) {
  v16bf f;
#pragma unroll
  for (int e = 0; e < 8; ++e) {
    f[e]     = (bf16_t)base[(long long)(kbase + e) * stride];
    f[8 + e] = (bf16_t)base[(long long)(kbase + 16 + e) * stride];
  }
  return f;
}

#define WMMA_BF16(a, b, c) \
  __builtin_amdgcn_wmma_f32_16x16x32_bf16(false, (a), false, (b), (short)0, (c), false, false)

// --------------------------- generic WMMA GEMM -----------------------------
// C[M,N] = A[M,K] * B[N,K]^T (+bias[N]) (+Cadd) (relu) ; optional bf16 copy.
// One wave per (16*BM x 16*BN) tile; A fragments reused BN times, B fragments
// reused BM times. grid = (N/(16*BN), M/(16*BM), batch).
template <int BM, int BN>
__global__ void gemm_bf16_wmma(const bf16_t* A, long long sAb, int lda,
                               const bf16_t* B, long long sBb, int ldb,
                               float* C, long long sCb, int ldc,
                               int K, const float* bias, int relu,
                               const float* Cadd, bf16_t* Cbf) {
  int lane = threadIdx.x & 31;
  int half = lane >> 4;
  int l15  = lane & 15;
  int tn = blockIdx.x, tm = blockIdx.y, bz = blockIdx.z;
  A += (long long)bz * sAb;
  B += (long long)bz * sBb;
  C += (long long)bz * sCb;
  if (Cadd) Cadd += (long long)bz * sCb;
  if (Cbf)  Cbf  += (long long)bz * sCb;

  const bf16_t* arow[BM];
  const bf16_t* brow[BN];
#pragma unroll
  for (int m = 0; m < BM; ++m)
    arow[m] = A + (long long)((tm * BM + m) * 16 + l15) * lda;
#pragma unroll
  for (int n = 0; n < BN; ++n)
    brow[n] = B + (long long)((tn * BN + n) * 16 + l15) * ldb;

  v8f acc[BM][BN] = {};
  for (int k0 = 0; k0 < K; k0 += 32) {
    int kbase = k0 + half * 8;
    v16bf af[BM];
    v16bf bfv[BN];
#pragma unroll
    for (int m = 0; m < BM; ++m) af[m] = frag_bf(arow[m], kbase);
#pragma unroll
    for (int n = 0; n < BN; ++n) bfv[n] = frag_bf(brow[n], kbase);
#pragma unroll
    for (int m = 0; m < BM; ++m)
#pragma unroll
      for (int n = 0; n < BN; ++n)
        acc[m][n] = WMMA_BF16(af[m], bfv[n], acc[m][n]);
  }

#pragma unroll
  for (int m = 0; m < BM; ++m) {
    int row0 = (tm * BM + m) * 16 + half * 8;
#pragma unroll
    for (int n = 0; n < BN; ++n) {
      int col = (tn * BN + n) * 16 + l15;
      float badd = bias ? bias[col] : 0.0f;
#pragma unroll
      for (int r = 0; r < 8; ++r) {
        long long idx = (long long)(row0 + r) * ldc + col;
        float v = acc[m][n][r] + badd;
        if (Cadd) v += Cadd[idx];
        if (relu) v = fmaxf(v, 0.0f);
        C[idx] = v;
        if (Cbf) Cbf[idx] = (bf16_t)v;
      }
    }
  }
}

// -------- pair bias: logits[h][i][j] = sqrt(1/3) * (p[i,j,:] @ wb[h,:]) ----
// (scale folded into wb_bf). grid = (Njtiles, NRES i). One wave per tile.
__global__ void pair_bias_kernel(const float* __restrict__ p,
                                 const bf16_t* __restrict__ wbf,
                                 float* __restrict__ logits) {
  int lane = threadIdx.x & 31;
  int half = lane >> 4;
  int l15  = lane & 15;
  int tm = blockIdx.x;
  int i  = blockIdx.y;
  const float*  arow = p + (long long)i * NRES * CPD + (long long)(tm * 16 + l15) * CPD;
  const bf16_t* brow = wbf + l15 * CPD;   // rows 12..15 are zero-padded
  __builtin_prefetch((const void*)(arow + 16 * CPD), 0, 0);
  v8f acc = {};
#pragma unroll
  for (int k0 = 0; k0 < CPD; k0 += 32) {
    v16bf af = frag_f32(arow, k0 + half * 8);
    v16bf bfv = frag_bf(brow, k0 + half * 8);
    acc = WMMA_BF16(af, bfv, acc);
  }
  int h = l15;
  if (h < NHD) {
    long long base = (long long)h * NNL + (long long)i * NRES + tm * 16 + half * 8;
#pragma unroll
    for (int r = 0; r < 8; ++r) logits[base + r] = acc[r];
  }
}

// ---- attention logits: one K=32 WMMA fuses qk-dot and point-cross terms ---
// grid = (jtiles, itiles, heads). logits updated in place (bias already there).
__global__ void attn_logit_kernel(const bf16_t* __restrict__ Abuf,
                                  const bf16_t* __restrict__ Bbuf,
                                  float* __restrict__ logits,
                                  const float* __restrict__ rq,
                                  const float* __restrict__ rk,
                                  const float* __restrict__ mask,
                                  const float* __restrict__ bbias) {
  int lane = threadIdx.x & 31;
  int half = lane >> 4;
  int l15  = lane & 15;
  int tn = blockIdx.x, tm = blockIdx.y, h = blockIdx.z;
  const bf16_t* arow = Abuf + ((long long)h * NRES + tm * 16 + l15) * 32;
  const bf16_t* brow = Bbuf + ((long long)h * NRES + tn * 16 + l15) * 32;
  v8f acc = {};
  acc = WMMA_BF16(frag_bf(arow, half * 8), frag_bf(brow, half * 8), acc);

  int j = tn * 16 + l15;
  float rkj = rk[h * NRES + j];
  float mj  = mask[j];
  float bb  = 0.57735026919f * bbias[h];  // sqrt(1/3) * bbias
  long long base = (long long)h * NNL;
#pragma unroll
  for (int r = 0; r < 8; ++r) {
    int i = tm * 16 + half * 8 + r;
    long long idx = base + (long long)i * NRES + j;
    float v = acc[r] + logits[idx] + rq[h * NRES + i] + rkj + bb +
              100000.0f * (mask[i] * mj - 1.0f);
    logits[idx] = v;
  }
}

// --------------------------- softmax over j --------------------------------
__global__ void softmax_kernel(const float* __restrict__ logits,
                               bf16_t* __restrict__ abf) {
  int i = blockIdx.x, h = blockIdx.y, t = threadIdx.x;
  const float* row = logits + (long long)h * NNL + (long long)i * NRES;
  bf16_t* orow = abf + (long long)h * NNL + (long long)i * NRES;
  __shared__ float red[256];
  float m = -1e30f;
  for (int j = t; j < NRES; j += 256) m = fmaxf(m, row[j]);
  red[t] = m; __syncthreads();
  for (int s = 128; s > 0; s >>= 1) { if (t < s) red[t] = fmaxf(red[t], red[t + s]); __syncthreads(); }
  m = red[0]; __syncthreads();
  float sum = 0.0f;
  for (int j = t; j < NRES; j += 256) sum += __expf(row[j] - m);
  red[t] = sum; __syncthreads();
  for (int s = 128; s > 0; s >>= 1) { if (t < s) red[t] += red[t + s]; __syncthreads(); }
  float inv = 1.0f / red[0];
  for (int j = t; j < NRES; j += 256) orow[j] = (bf16_t)(__expf(row[j] - m) * inv);
}

// ------- opair[i][h][z] = sum_j a[h,i,j] * p[i,j,z]  (per-i WMMA GEMM) -----
// grid = (4 ztile-pairs, NRES i). A rows = heads (planes 12..15 zero).
// Two z-tiles per wave: A fragment reused, halves abf traffic.
__global__ void opair_kernel(const bf16_t* __restrict__ abf,
                             const float* __restrict__ p,
                             float* __restrict__ opair) {
  int lane = threadIdx.x & 31;
  int half = lane >> 4;
  int l15  = lane & 15;
  int tn = blockIdx.x;
  int i  = blockIdx.y;
  const bf16_t* arow   = abf + (long long)l15 * NNL + (long long)i * NRES;
  const float*  bbase0 = p + (long long)i * NRES * CPD + tn * 32 + l15;
  const float*  bbase1 = bbase0 + 16;
  v8f acc0 = {}, acc1 = {};
  for (int k0 = 0; k0 < NRES; k0 += 32) {
    if (k0 + 32 < NRES)
      __builtin_prefetch((const void*)(bbase0 + (long long)(k0 + 32) * CPD), 0, 0);
    int kbase = k0 + half * 8;
    v16bf af = frag_bf(arow, kbase);
    acc0 = WMMA_BF16(af, frag_f32_strided(bbase0, kbase, CPD), acc0);
    acc1 = WMMA_BF16(af, frag_f32_strided(bbase1, kbase, CPD), acc1);
  }
  int z0 = tn * 32 + l15;
#pragma unroll
  for (int r = 0; r < 8; ++r) {
    int h = half * 8 + r;
    opair[(long long)i * 2048 + h * CPD + z0]      = acc0[r];
    opair[(long long)i * 2048 + h * CPD + z0 + 16] = acc1[r];
  }
}

// ---- unpack projections, rotate points, build augmented attn operands -----
// Abuf[h][i][0:16]=q*sA, [16:28]=qp*hw ; Bbuf: k, kp ; rq/rk = -0.5*hw*|pts|^2
__global__ void rot_unpack_kernel(const float* __restrict__ proj,
                                  const float* __restrict__ rots,
                                  const float* __restrict__ trans,
                                  const float* __restrict__ head_w,
                                  bf16_t* __restrict__ Abuf, bf16_t* __restrict__ Bbuf,
                                  float* __restrict__ rq, float* __restrict__ rk,
                                  bf16_t* __restrict__ vTbf, bf16_t* __restrict__ vpTbf) {
  int i = blockIdx.x, t = threadIdx.x;
  __shared__ float R[9], T[3], q2s[NHD], k2s[NHD], hws[NHD];
  if (t < 9) R[t] = rots[i * 9 + t];
  if (t < 3) T[t] = trans[i * 3 + t];
  if (t < NHD) {
    q2s[t] = 0.0f; k2s[t] = 0.0f;
    float x = head_w[t];
    float sp = (x > 20.0f) ? x : log1pf(__expf(x));
    hws[t] = sp * sqrtf(1.0f / 54.0f);  // sqrt(1/(3*(PQ*9/2)))
  }
  __syncthreads();
  const float* pr = proj + (long long)i * PROJC;
  const float sA = 0.14433756730f;  // sqrt(1/(3*CH)) = sqrt(1/48)

  if (t < 192) {  // q / k / v
    int h = t >> 4, c = t & 15;
    Abuf[((long long)h * NRES + i) * 32 + c] = (bf16_t)(pr[t] * sA);
    Bbuf[((long long)h * NRES + i) * 32 + c] = (bf16_t)(pr[192 + h * 32 + c]);
    vTbf[((long long)h * 16 + c) * NRES + i] = (bf16_t)(pr[192 + h * 32 + 16 + c]);
  }
  if (t < 48) {  // query points
    int h = t / PQD, pp = t % PQD;
    float y0 = pr[576 + t], y1 = pr[576 + 48 + t], y2 = pr[576 + 96 + t];
    float vx = R[0] * y0 + R[1] * y1 + R[2] * y2 + T[0];
    float vy = R[3] * y0 + R[4] * y1 + R[5] * y2 + T[1];
    float vz = R[6] * y0 + R[7] * y1 + R[8] * y2 + T[2];
    float hw = hws[h];
    long long base = ((long long)h * NRES + i) * 32 + 16 + pp * 3;
    Abuf[base]     = (bf16_t)(vx * hw);
    Abuf[base + 1] = (bf16_t)(vy * hw);
    Abuf[base + 2] = (bf16_t)(vz * hw);
    atomicAdd(&q2s[h], hw * (vx * vx + vy * vy + vz * vz));
  }
  if (t < 144) {  // key + value points
    int h = t / 12, pp = t % 12;
    float y0 = pr[720 + t], y1 = pr[720 + 144 + t], y2 = pr[720 + 288 + t];
    float vx = R[0] * y0 + R[1] * y1 + R[2] * y2 + T[0];
    float vy = R[3] * y0 + R[4] * y1 + R[5] * y2 + T[1];
    float vz = R[6] * y0 + R[7] * y1 + R[8] * y2 + T[2];
    if (pp < PQD) {
      long long base = ((long long)h * NRES + i) * 32 + 16 + pp * 3;
      Bbuf[base]     = (bf16_t)vx;
      Bbuf[base + 1] = (bf16_t)vy;
      Bbuf[base + 2] = (bf16_t)vz;
      atomicAdd(&k2s[h], hws[h] * (vx * vx + vy * vy + vz * vz));
    } else {
      int pv = pp - PQD;
      vpTbf[((long long)h * 32 + pv * 3 + 0) * NRES + i] = (bf16_t)vx;
      vpTbf[((long long)h * 32 + pv * 3 + 1) * NRES + i] = (bf16_t)vy;
      vpTbf[((long long)h * 32 + pv * 3 + 2) * NRES + i] = (bf16_t)vz;
    }
  }
  __syncthreads();
  if (t < NHD) {
    rq[t * NRES + i] = -0.5f * q2s[t];
    rk[t * NRES + i] = -0.5f * k2s[t];
  }
}

// ------- cat = [o, op_x, op_y, op_z, |op|, opair]  (inverse-rotated) -------
__global__ void cat_kernel(const float* __restrict__ obuf,
                           const float* __restrict__ opbuf,
                           const float* __restrict__ opairb,
                           const float* __restrict__ rots,
                           const float* __restrict__ trans,
                           bf16_t* __restrict__ catbf) {
  int i = blockIdx.x, t = threadIdx.x;
  __shared__ float R[9], T[3];
  if (t < 9) R[t] = rots[i * 9 + t];
  if (t < 3) T[t] = trans[i * 3 + t];
  __syncthreads();
  bf16_t* crow = catbf + (long long)i * CATC;
  for (int c = t; c < 192; c += 256) crow[c] = (bf16_t)obuf[(long long)i * 192 + c];
  if (t < 96) {
    int h = t / PVD, pv = t % PVD;
    const float* op = opbuf + (long long)i * 384 + h * 32 + pv * 3;
    float x = op[0] - T[0], y = op[1] - T[1], z = op[2] - T[2];
    float lx = R[0] * x + R[3] * y + R[6] * z;  // R^T
    float ly = R[1] * x + R[4] * y + R[7] * z;
    float lz = R[2] * x + R[5] * y + R[8] * z;
    crow[192 + t] = (bf16_t)lx;
    crow[288 + t] = (bf16_t)ly;
    crow[384 + t] = (bf16_t)lz;
    crow[480 + t] = (bf16_t)sqrtf(lx * lx + ly * ly + lz * lz + 1e-8f);
  }
  for (int c = t; c < 1536; c += 256)
    crow[576 + c] = (bf16_t)opairb[(long long)i * 2048 + c];
}

// ------------------------------ layernorm ----------------------------------
__global__ void layernorm_kernel(const float* __restrict__ x,
                                 const float* __restrict__ g,
                                 const float* __restrict__ b,
                                 float* __restrict__ y, bf16_t* __restrict__ ybf) {
  int i = blockIdx.x, t = threadIdx.x;
  const float* row = x + (long long)i * CSD;
  __shared__ float red[128];
  float s = 0.0f;
  for (int c = t; c < CSD; c += 128) s += row[c];
  red[t] = s; __syncthreads();
  for (int k = 64; k > 0; k >>= 1) { if (t < k) red[t] += red[t + k]; __syncthreads(); }
  float mean = red[0] / CSD; __syncthreads();
  float v = 0.0f;
  for (int c = t; c < CSD; c += 128) { float d = row[c] - mean; v += d * d; }
  red[t] = v; __syncthreads();
  for (int k = 64; k > 0; k >>= 1) { if (t < k) red[t] += red[t + k]; __syncthreads(); }
  float inv = rsqrtf(red[0] / CSD + 1e-5f);
  for (int c = t; c < CSD; c += 128) {
    float o = (row[c] - mean) * inv * g[c] + b[c];
    y[(long long)i * CSD + c] = o;
    if (ybf) ybf[(long long)i * CSD + c] = (bf16_t)o;
  }
}

// ----------------- frame update: quat -> rot, compose ----------------------
__global__ void pose_kernel(const float* __restrict__ sfin,
                            const float* __restrict__ wbb,
                            const float* __restrict__ bbb,
                            const float* __restrict__ rots,
                            const float* __restrict__ trans,
                            float* __restrict__ rout, float* __restrict__ tout) {
  int i = blockIdx.x, lane = threadIdx.x;
  const float* row = sfin + (long long)i * CSD;
  float acc[6] = {0, 0, 0, 0, 0, 0};
  for (int k = lane; k < CSD; k += 32) {
    float sv = row[k];
#pragma unroll
    for (int o = 0; o < 6; ++o) acc[o] += sv * wbb[o * CSD + k];
  }
#pragma unroll
  for (int off = 16; off > 0; off >>= 1)
#pragma unroll
    for (int o = 0; o < 6; ++o) acc[o] += __shfl_down(acc[o], off, 32);
  if (lane == 0) {
    float u[6];
#pragma unroll
    for (int o = 0; o < 6; ++o) u[o] = acc[o] + bbb[o];
    float qx = u[0], qy = u[1], qz = u[2];
    float n = rsqrtf(1.0f + qx * qx + qy * qy + qz * qz);
    float w = n, x = qx * n, y = qy * n, z = qz * n;
    float U[9] = {w*w + x*x - y*y - z*z, 2*(x*y - w*z),         2*(x*z + w*y),
                  2*(x*y + w*z),         w*w - x*x + y*y - z*z, 2*(y*z - w*x),
                  2*(x*z - w*y),         2*(y*z + w*x),         w*w - x*x - y*y + z*z};
    const float* R = rots + i * 9;
#pragma unroll
    for (int a = 0; a < 3; ++a)
#pragma unroll
      for (int c = 0; c < 3; ++c) {
        float s2 = 0.0f;
#pragma unroll
        for (int b = 0; b < 3; ++b) s2 += R[a * 3 + b] * U[b * 3 + c];
        rout[i * 9 + a * 3 + c] = s2;
      }
#pragma unroll
    for (int a = 0; a < 3; ++a)
      tout[i * 3 + a] = R[a * 3 + 0] * u[3] + R[a * 3 + 1] * u[4] + R[a * 3 + 2] * u[5] +
                        trans[i * 3 + a];
  }
}

// ------------------------------ conversions --------------------------------
__global__ void cvt_f32_bf16_kernel(const float* __restrict__ src, bf16_t* __restrict__ dst,
                                    long long n) {
  long long idx = (long long)blockIdx.x * blockDim.x + threadIdx.x;
  long long stride = (long long)gridDim.x * blockDim.x;
  for (; idx < n; idx += stride) dst[idx] = (bf16_t)src[idx];
}

__global__ void scale_cvt_kernel(const float* __restrict__ src, bf16_t* __restrict__ dst,
                                 float scale, long long n) {
  long long idx = (long long)blockIdx.x * blockDim.x + threadIdx.x;
  long long stride = (long long)gridDim.x * blockDim.x;
  for (; idx < n; idx += stride) dst[idx] = (bf16_t)(src[idx] * scale);
}

// ---------------------------------------------------------------------------
static inline size_t al256(size_t x) { return (x + 255) & ~(size_t)255; }

extern "C" void kernel_launch(void* const* d_in, const int* in_sizes, int n_in,
                              void* d_out, int out_size, void* d_ws, size_t ws_size,
                              hipStream_t stream) {
  (void)in_sizes; (void)n_in; (void)out_size; (void)ws_size;
  const float* s_in   = (const float*)d_in[0];
  const float* p_in   = (const float*)d_in[1];
  const float* rots   = (const float*)d_in[2];
  const float* trans  = (const float*)d_in[3];
  const float* mask   = (const float*)d_in[4];
  const float* wq     = (const float*)d_in[5];
  const float* bq     = (const float*)d_in[6];
  const float* wkv    = (const float*)d_in[7];
  const float* bkv    = (const float*)d_in[8];
  const float* wqp    = (const float*)d_in[9];
  const float* bqp    = (const float*)d_in[10];
  const float* wkvp   = (const float*)d_in[11];
  const float* bkvp   = (const float*)d_in[12];
  const float* wb     = (const float*)d_in[13];
  const float* bbias  = (const float*)d_in[14];
  const float* head_w = (const float*)d_in[15];
  const float* wo     = (const float*)d_in[16];
  const float* bo     = (const float*)d_in[17];
  const float* ln1_g  = (const float*)d_in[18];
  const float* ln1_b  = (const float*)d_in[19];
  const float* t_w1   = (const float*)d_in[20];
  const float* t_b1   = (const float*)d_in[21];
  const float* t_w2   = (const float*)d_in[22];
  const float* t_b2   = (const float*)d_in[23];
  const float* t_w3   = (const float*)d_in[24];
  const float* t_b3   = (const float*)d_in[25];
  const float* ln2_g  = (const float*)d_in[26];
  const float* ln2_b  = (const float*)d_in[27];
  const float* wbb    = (const float*)d_in[28];
  const float* bbb    = (const float*)d_in[29];

  // ---------------- workspace carve-up (~121 MB) ----------------
  char* w = (char*)d_ws;
  size_t off = 0;
  auto carve = [&](size_t bytes) { char* p = w + off; off += al256(bytes); return p; };

  bf16_t* sbf    = (bf16_t*)carve((size_t)NRES * CSD * 2);
  bf16_t* Wbig   = (bf16_t*)carve((size_t)PROJC * CSD * 2);
  float*  bbig   = (float*) carve((size_t)PROJC * 4);
  float*  proj   = (float*) carve((size_t)NRES * PROJC * 4);
  bf16_t* Abuf   = (bf16_t*)carve((size_t)NHD * NRES * 32 * 2);
  bf16_t* Bbuf   = (bf16_t*)carve((size_t)NHD * NRES * 32 * 2);
  float*  rq     = (float*) carve((size_t)NHD * NRES * 4);
  float*  rk     = (float*) carve((size_t)NHD * NRES * 4);
  bf16_t* vTbf   = (bf16_t*)carve((size_t)NHD * 16 * NRES * 2);
  bf16_t* vpTbf  = (bf16_t*)carve((size_t)NHD * 32 * NRES * 2);
  bf16_t* wbbf   = (bf16_t*)carve((size_t)16 * CPD * 2);
  float*  logits = (float*) carve((size_t)NHD * NNL * 4);
  bf16_t* abf    = (bf16_t*)carve((size_t)16 * NNL * 2);   // 16 planes (12 used)
  float*  obuf   = (float*) carve((size_t)NRES * 192 * 4);
  float*  opbuf  = (float*) carve((size_t)NRES * 384 * 4);
  float*  opairb = (float*) carve((size_t)NRES * 2048 * 4);
  bf16_t* catbf  = (bf16_t*)carve((size_t)NRES * CATC * 2);
  bf16_t* wobf   = (bf16_t*)carve((size_t)CSD * CATC * 2);
  float*  sres   = (float*) carve((size_t)NRES * CSD * 4);
  float*  sm1    = (float*) carve((size_t)NRES * CSD * 4);
  bf16_t* s1bf   = (bf16_t*)carve((size_t)NRES * CSD * 2);
  bf16_t* tw1bf  = (bf16_t*)carve((size_t)CSD * CSD * 2);
  bf16_t* tw2bf  = (bf16_t*)carve((size_t)CSD * CSD * 2);
  bf16_t* tw3bf  = (bf16_t*)carve((size_t)CSD * CSD * 2);
  float*  h1     = (float*) carve((size_t)NRES * CSD * 4);
  bf16_t* h1bf   = (bf16_t*)carve((size_t)NRES * CSD * 2);
  float*  h2     = (float*) carve((size_t)NRES * CSD * 4);
  bf16_t* h2bf   = (bf16_t*)carve((size_t)NRES * CSD * 2);
  float*  s2buf  = (float*) carve((size_t)NRES * CSD * 4);

  float* s_out   = (float*)d_out;                               // 1024*384
  float* rots_o  = s_out + (size_t)NRES * CSD;                  // 1024*9
  float* trans_o = rots_o + (size_t)NRES * 9;                   // 1024*3

  // ---------------- padding invariants (every call) ----------------
  hipMemsetAsync(Abuf,  0, (size_t)NHD * NRES * 32 * 2, stream);
  hipMemsetAsync(Bbuf,  0, (size_t)NHD * NRES * 32 * 2, stream);
  hipMemsetAsync(vpTbf, 0, (size_t)NHD * 32 * NRES * 2, stream);
  hipMemsetAsync(wbbf,  0, (size_t)16 * CPD * 2, stream);
  hipMemsetAsync(abf,   0, (size_t)16 * NNL * 2, stream);

  // ---------------- weight/activation conversions ----------------
  auto cvt = [&](const float* src, bf16_t* dst, long long n) {
    int blocks = (int)((n + 255) / 256); if (blocks > 4096) blocks = 4096;
    cvt_f32_bf16_kernel<<<blocks, 256, 0, stream>>>(src, dst, n);
  };
  cvt(s_in, sbf, (long long)NRES * CSD);
  cvt(wq,   Wbig,                   (long long)192 * CSD);
  cvt(wkv,  Wbig + 192 * CSD,       (long long)384 * CSD);
  cvt(wqp,  Wbig + 576 * CSD,       (long long)144 * CSD);
  cvt(wkvp, Wbig + 720 * CSD,       (long long)432 * CSD);
  cvt(wo,   wobf, (long long)CSD * CATC);
  cvt(t_w1, tw1bf, (long long)CSD * CSD);
  cvt(t_w2, tw2bf, (long long)CSD * CSD);
  cvt(t_w3, tw3bf, (long long)CSD * CSD);
  scale_cvt_kernel<<<8, 256, 0, stream>>>(wb, wbbf, 0.57735026919f, (long long)NHD * CPD);

  hipMemcpyAsync(bbig,        bq,   192 * 4, hipMemcpyDeviceToDevice, stream);
  hipMemcpyAsync(bbig + 192,  bkv,  384 * 4, hipMemcpyDeviceToDevice, stream);
  hipMemcpyAsync(bbig + 576,  bqp,  144 * 4, hipMemcpyDeviceToDevice, stream);
  hipMemcpyAsync(bbig + 720,  bkvp, 432 * 4, hipMemcpyDeviceToDevice, stream);

  // 1) fused input projections: proj = s @ [wq|wkv|wqp|wkvp]^T + biases
  gemm_bf16_wmma<2, 4><<<dim3(PROJC / 64, NRES / 32, 1), 32, 0, stream>>>(
      sbf, 0, CSD, Wbig, 0, CSD, proj, 0, PROJC, CSD, bbig, 0, nullptr, nullptr);

  // 2) rotate points, build augmented attention operands, v / vp transposes
  rot_unpack_kernel<<<NRES, 192, 0, stream>>>(proj, rots, trans, head_w,
                                              Abuf, Bbuf, rq, rk, vTbf, vpTbf);

  // 3) pair bias (p pass #1) directly into logits (pre-scaled)
  pair_bias_kernel<<<dim3(NRES / 16, NRES), 32, 0, stream>>>(p_in, wbbf, logits);

  // 4) qk + point-attention logits, fused in one WMMA per tile
  attn_logit_kernel<<<dim3(NRES / 16, NRES / 16, NHD), 32, 0, stream>>>(
      Abuf, Bbuf, logits, rq, rk, mask, bbias);

  // 5) softmax -> bf16 attention weights
  softmax_kernel<<<dim3(NRES, NHD), 256, 0, stream>>>(logits, abf);

  // 6) o = a @ v   (batched over heads, N=16)
  gemm_bf16_wmma<2, 1><<<dim3(1, NRES / 32, NHD), 32, 0, stream>>>(
      abf, NNL, NRES, vTbf, 16LL * NRES, NRES, obuf, 16, 192, NRES,
      nullptr, 0, nullptr, nullptr);

  // 7) op = a @ vp (batched over heads, 24 cols padded to 32)
  gemm_bf16_wmma<2, 2><<<dim3(1, NRES / 32, NHD), 32, 0, stream>>>(
      abf, NNL, NRES, vpTbf, 32LL * NRES, NRES, opbuf, 32, 384, NRES,
      nullptr, 0, nullptr, nullptr);

  // 8) opair = a @ p (p pass #2, per-residue WMMA GEMMs, 2 z-tiles/wave)
  opair_kernel<<<dim3(CPD / 32, NRES), 32, 0, stream>>>(abf, p_in, opairb);

  // 9) inverse-rotate op, norms, concatenate
  cat_kernel<<<NRES, 256, 0, stream>>>(obuf, opbuf, opairb, rots, trans, catbf);

  // 10) output projection + residual
  gemm_bf16_wmma<2, 4><<<dim3(CSD / 64, NRES / 32, 1), 32, 0, stream>>>(
      catbf, 0, CATC, wobf, 0, CATC, sres, 0, CSD, CATC, bo, 0, s_in, nullptr);

  // 11) layernorm 1
  layernorm_kernel<<<NRES, 128, 0, stream>>>(sres, ln1_g, ln1_b, sm1, s1bf);

  // 12) transition MLP
  gemm_bf16_wmma<2, 4><<<dim3(CSD / 64, NRES / 32, 1), 32, 0, stream>>>(
      s1bf, 0, CSD, tw1bf, 0, CSD, h1, 0, CSD, CSD, t_b1, 1, nullptr, h1bf);
  gemm_bf16_wmma<2, 4><<<dim3(CSD / 64, NRES / 32, 1), 32, 0, stream>>>(
      h1bf, 0, CSD, tw2bf, 0, CSD, h2, 0, CSD, CSD, t_b2, 1, nullptr, h2bf);
  gemm_bf16_wmma<2, 4><<<dim3(CSD / 64, NRES / 32, 1), 32, 0, stream>>>(
      h2bf, 0, CSD, tw3bf, 0, CSD, s2buf, 0, CSD, CSD, t_b3, 0, sm1, nullptr);

  // 13) layernorm 2 -> s output
  layernorm_kernel<<<NRES, 128, 0, stream>>>(s2buf, ln2_g, ln2_b, s_out, nullptr);

  // 14) backbone frame update -> rots_new, trans_new
  pose_kernel<<<NRES, 32, 0, stream>>>(s_out, wbb, bbb, rots, trans, rots_o, trans_o);
}
